// MultiHeadAttentionWithCache_7773890806357
// MI455X (gfx1250) — compile-verified, matrix operand order checked by reading, not run on hardware
//
#include <hip/hip_runtime.h>
#include <hip/hip_bf16.h>

// MI455X (gfx1250) wave32 WMMA implementation of MHA-with-cache.
// All GEMM stages use v_wmma_f32_16x16x32_bf16 with bf16 hi/lo (bf16x3) split
// for near-fp32 accuracy; attention is flash-style with online softmax.

#define D_MODEL 1024
#define N_HEAD  16
#define HDIM    64
#define BATCH   4
#define TLEN    1024
#define SCACHE  3072
#define SLEN    (SCACHE + TLEN)   // 4096

typedef __attribute__((ext_vector_type(16))) __bf16 v16bf;
typedef __attribute__((ext_vector_type(8)))  float  v8f;

union Frag {
    v16bf v;
    unsigned short s[16];
    unsigned u[8];
};

__device__ __forceinline__ unsigned short f2bf(float f) {
    union { float f; unsigned u; } c; c.f = f;
    unsigned u = c.u;
    u += 0x7fffu + ((u >> 16) & 1u);          // round to nearest even
    return (unsigned short)(u >> 16);
}
__device__ __forceinline__ float bf2f(unsigned short h) {
    union { unsigned u; float f; } c; c.u = ((unsigned)h) << 16;
    return c.f;
}

// A/B fragment (16x32 bf16) element k-offset for lane half `half`:
//   k(i) = kstep + half*8 + i + (i>=8 ? 8 : 0)
// Row-major gather: element (idx, k) at p0[idx*ld + k]  (A tiles, B stored [n][k])
__device__ __forceinline__ v16bf frag_rowmajor(const unsigned short* p0, int ld,
                                               int idx, int half, int kstep) {
    const unsigned short* p = p0 + idx * ld + kstep + half * 8;
    Frag f;
#pragma unroll
    for (int j = 0; j < 4; ++j) f.u[j]     = *(const unsigned*)(p + 2 * j);
#pragma unroll
    for (int j = 0; j < 4; ++j) f.u[4 + j] = *(const unsigned*)(p + 16 + 2 * j);
    return f.v;
}

// k-major gather: element (k, idx) at p0[k*ld + idx]  (B stored [k][n], e.g. V tile)
__device__ __forceinline__ v16bf frag_kmajor(const unsigned short* p0, int ld,
                                             int idx, int half, int kstep) {
    Frag f;
#pragma unroll
    for (int i = 0; i < 16; ++i) {
        int k = kstep + half * 8 + i + (i >= 8 ? 8 : 0);
        f.s[i] = p0[k * ld + idx];
    }
    return f.v;
}

// bf16x3: C += Ahi*Bhi + Ahi*Blo + Alo*Bhi  (f32 accumulate)
__device__ __forceinline__ v8f wmma3(v16bf ah, v16bf al, v16bf bh, v16bf bl, v8f c) {
    c = __builtin_amdgcn_wmma_f32_16x16x32_bf16(false, al, false, bh, (short)0, c, false, false);
    c = __builtin_amdgcn_wmma_f32_16x16x32_bf16(false, ah, false, bl, (short)0, c, false, false);
    c = __builtin_amdgcn_wmma_f32_16x16x32_bf16(false, ah, false, bh, (short)0, c, false, false);
    return c;
}

// ---------------------------------------------------------------------------
// GEMM: out[M=4096 x N=1024] = A[4096x1024] @ W[1024x1024] (+bias)
// Block tile 128x64, 8 waves in 4x2 grid, each wave 32x32 (four 16x16 WMMAs).
// kvmode: remap output row (b*T + t) -> (b*S + SCACHE + t) for K/V segments.
// ---------------------------------------------------------------------------
__global__ __launch_bounds__(256)
void gemm_bf16x3(const float* __restrict__ A, const float* __restrict__ W,
                 const float* __restrict__ bias, float* __restrict__ out, int kvmode) {
    __shared__ __align__(16) unsigned short Ah[128 * 32], Al[128 * 32];
    __shared__ __align__(16) unsigned short Bh[64 * 32],  Bl[64 * 32];   // stored [n][k]

    const int tid = threadIdx.x, lane = tid & 31, wid = tid >> 5;
    const int idx = lane & 15, half = lane >> 4;
    const int wm = (wid & 3) * 32, wn = (wid >> 2) * 32;
    const int m0 = blockIdx.x * 128, n0 = blockIdx.y * 64;

    v8f acc[2][2] = {};

    for (int kt = 0; kt < D_MODEL; kt += 32) {
        __syncthreads();
#pragma unroll
        for (int it = 0; it < 16; ++it) {               // 128x32 A tile
            int e = tid + it * 256; int r = e >> 5, c = e & 31;
            float f = A[(m0 + r) * D_MODEL + kt + c];
            unsigned short h = f2bf(f);
            Ah[e] = h; Al[e] = f2bf(f - bf2f(h));
        }
#pragma unroll
        for (int it = 0; it < 8; ++it) {                // 32x64 W tile -> [n][k]
            int e = tid + it * 256; int r = e >> 6, c = e & 63;
            float f = W[(kt + r) * D_MODEL + n0 + c];
            unsigned short h = f2bf(f);
            Bh[c * 32 + r] = h; Bl[c * 32 + r] = f2bf(f - bf2f(h));
        }
        __syncthreads();
#pragma unroll
        for (int tm = 0; tm < 2; ++tm) {
            v16bf ah = frag_rowmajor(Ah + (wm + tm * 16) * 32, 32, idx, half, 0);
            v16bf al = frag_rowmajor(Al + (wm + tm * 16) * 32, 32, idx, half, 0);
#pragma unroll
            for (int tn = 0; tn < 2; ++tn) {
                v16bf bh = frag_rowmajor(Bh + (wn + tn * 16) * 32, 32, idx, half, 0);
                v16bf bl = frag_rowmajor(Bl + (wn + tn * 16) * 32, 32, idx, half, 0);
                acc[tm][tn] = wmma3(ah, al, bh, bl, acc[tm][tn]);
            }
        }
    }

#pragma unroll
    for (int tm = 0; tm < 2; ++tm)
#pragma unroll
        for (int tn = 0; tn < 2; ++tn)
#pragma unroll
            for (int e = 0; e < 8; ++e) {
                int row = m0 + wm + tm * 16 + e + half * 8;
                int col = n0 + wn + tn * 16 + idx;
                float v = acc[tm][tn][e] + (bias ? bias[col] : 0.0f);
                long orow = kvmode ? ((long)(row >> 10) * SLEN + SCACHE + (row & 1023))
                                   : (long)row;
                out[orow * D_MODEL + col] = v;
            }
}

// ---------------------------------------------------------------------------
// Flash attention. Grid (T/128, H, B), 256 threads (8 waves).
// Each wave owns 16 query rows; block stages 32-row K/V chunks in LDS.
// Mask (faithful to ref): position s masked iff s < T && s > t_query.
// ---------------------------------------------------------------------------
__global__ __launch_bounds__(256)
void attn_flash(const float* __restrict__ Q, const float* __restrict__ K,
                const float* __restrict__ V, float* __restrict__ O) {
    __shared__ __align__(16) unsigned short Kh[32 * 64], Kl[32 * 64];   // [s][k]
    __shared__ __align__(16) unsigned short Vh[32 * 64], Vl[32 * 64];   // [s][hd]
    __shared__ __align__(16) unsigned short Ph[8][16 * 32], Pl[8][16 * 32];

    const int tid = threadIdx.x, lane = tid & 31, wid = tid >> 5;
    const int idx = lane & 15, half = lane >> 4;
    const int b = blockIdx.z, h = blockIdx.y;
    const int t0 = blockIdx.x * 128 + wid * 16;       // wave's first query row

    // Q fragments (16 rows x 64 k), scale = hd^-0.5 = 0.125 folded into q
    Frag qhF[2], qlF[2];
#pragma unroll
    for (int ks = 0; ks < 2; ++ks)
#pragma unroll
        for (int i = 0; i < 16; ++i) {
            int k = ks * 32 + half * 8 + i + (i >= 8 ? 8 : 0);
            float f = Q[((long)(b * TLEN + t0 + idx)) * D_MODEL + h * HDIM + k] * 0.125f;
            unsigned short hb = f2bf(f);
            qhF[ks].s[i] = hb; qlF[ks].s[i] = f2bf(f - bf2f(hb));
        }

    float mr[8], lr[8];
    v8f o[4] = {};
#pragma unroll
    for (int e = 0; e < 8; ++e) { mr[e] = -1e30f; lr[e] = 0.0f; }

    for (int sc = 0; sc < SLEN; sc += 32) {
        __syncthreads();
#pragma unroll
        for (int it = 0; it < 8; ++it) {              // 32x64 K and V chunks
            int e = tid + it * 256; int s = e >> 6, k = e & 63;
            long gi = ((long)(b * SLEN + sc + s)) * D_MODEL + h * HDIM + k;
            float fk = K[gi]; unsigned short hb = f2bf(fk);
            Kh[e] = hb; Kl[e] = f2bf(fk - bf2f(hb));
            float fv = V[gi]; hb = f2bf(fv);
            Vh[e] = hb; Vl[e] = f2bf(fv - bf2f(hb));
        }
        __syncthreads();

        // S = Q K^T : 16 rows x 32 cols (two 16x16 C tiles)
        v8f sf[2] = {};
#pragma unroll
        for (int ng = 0; ng < 2; ++ng)
#pragma unroll
            for (int ks = 0; ks < 2; ++ks) {
                v16bf bh = frag_rowmajor(Kh + (ng * 16) * 64, 64, idx, half, ks * 32);
                v16bf bl = frag_rowmajor(Kl + (ng * 16) * 64, 64, idx, half, ks * 32);
                sf[ng] = wmma3(qhF[ks].v, qlF[ks].v, bh, bl, sf[ng]);
            }

        // mask + per-chunk row max
        float mc[8];
#pragma unroll
        for (int e = 0; e < 8; ++e) {
            int tq = t0 + e + half * 8;
            float a0 = sf[0][e]; int s0 = sc + idx;
            if (s0 < TLEN && s0 > tq) a0 = -1e30f;
            float a1 = sf[1][e]; int s1 = sc + 16 + idx;
            if (s1 < TLEN && s1 > tq) a1 = -1e30f;
            sf[0][e] = a0; sf[1][e] = a1;
            mc[e] = fmaxf(a0, a1);
        }
#pragma unroll
        for (int off = 1; off < 16; off <<= 1)
#pragma unroll
            for (int e = 0; e < 8; ++e) mc[e] = fmaxf(mc[e], __shfl_xor(mc[e], off, 32));

        float alpha[8];
#pragma unroll
        for (int e = 0; e < 8; ++e) {
            float mn = fmaxf(mr[e], mc[e]);
            alpha[e] = __expf(mr[e] - mn);
            mr[e] = mn;
        }

        // P = exp(S - m), stage to wave-private LDS in [row][col] order
        float rs[8];
#pragma unroll
        for (int e = 0; e < 8; ++e) {
            float pa = (sf[0][e] <= -1e29f) ? 0.0f : __expf(sf[0][e] - mr[e]);
            float pb = (sf[1][e] <= -1e29f) ? 0.0f : __expf(sf[1][e] - mr[e]);
            int row = e + half * 8;
            unsigned short hb = f2bf(pa);
            Ph[wid][row * 32 + idx] = hb;      Pl[wid][row * 32 + idx]      = f2bf(pa - bf2f(hb));
            hb = f2bf(pb);
            Ph[wid][row * 32 + 16 + idx] = hb; Pl[wid][row * 32 + 16 + idx] = f2bf(pb - bf2f(hb));
            rs[e] = pa + pb;
        }
#pragma unroll
        for (int off = 1; off < 16; off <<= 1)
#pragma unroll
            for (int e = 0; e < 8; ++e) rs[e] += __shfl_xor(rs[e], off, 32);
#pragma unroll
        for (int e = 0; e < 8; ++e) lr[e] = lr[e] * alpha[e] + rs[e];

        asm volatile("s_wait_dscnt 0" ::: "memory");   // wave-local P store->load

        v16bf pah = frag_rowmajor(Ph[wid], 32, idx, half, 0);
        v16bf pal = frag_rowmajor(Pl[wid], 32, idx, half, 0);
#pragma unroll
        for (int g = 0; g < 4; ++g) {                  // O[16 x 64] in 4 C tiles
#pragma unroll
            for (int e = 0; e < 8; ++e) o[g][e] *= alpha[e];
            v16bf vh = frag_kmajor(Vh + g * 16, 64, idx, half, 0);
            v16bf vl = frag_kmajor(Vl + g * 16, 64, idx, half, 0);
            o[g] = wmma3(pah, pal, vh, vl, o[g]);
        }
    }

#pragma unroll
    for (int g = 0; g < 4; ++g)
#pragma unroll
        for (int e = 0; e < 8; ++e) {
            int tq = t0 + e + half * 8;
            O[((long)(b * TLEN + tq)) * D_MODEL + h * HDIM + g * 16 + idx] = o[g][e] / lr[e];
        }
}

extern "C" void kernel_launch(void* const* d_in, const int* in_sizes, int n_in,
                              void* d_out, int out_size, void* d_ws, size_t ws_size,
                              hipStream_t stream) {
    (void)in_sizes; (void)n_in; (void)out_size; (void)ws_size;
    const float* x  = (const float*)d_in[0];
    const float* kc = (const float*)d_in[1];
    const float* vc = (const float*)d_in[2];
    const float* Wq = (const float*)d_in[3];
    const float* bq = (const float*)d_in[4];
    const float* Wk = (const float*)d_in[5];
    const float* Wv = (const float*)d_in[6];
    const float* bv = (const float*)d_in[7];
    const float* Wo = (const float*)d_in[8];
    const float* bo = (const float*)d_in[9];

    float* outp = (float*)d_out;
    float* kseg = outp + (size_t)BATCH * TLEN * D_MODEL;
    float* vseg = kseg + (size_t)BATCH * SLEN * D_MODEL;

    float* qws    = (float*)d_ws;                                  // 16 MB
    float* attnws = qws + (size_t)BATCH * TLEN * D_MODEL;          // 16 MB

    // caches -> first SCACHE rows of each batch's K/V segment
    for (int b = 0; b < BATCH; ++b) {
        hipMemcpyAsync(kseg + (size_t)b * SLEN * D_MODEL, kc + (size_t)b * SCACHE * D_MODEL,
                       (size_t)SCACHE * D_MODEL * sizeof(float), hipMemcpyDeviceToDevice, stream);
        hipMemcpyAsync(vseg + (size_t)b * SLEN * D_MODEL, vc + (size_t)b * SCACHE * D_MODEL,
                       (size_t)SCACHE * D_MODEL * sizeof(float), hipMemcpyDeviceToDevice, stream);
    }

    dim3 ggrid(BATCH * TLEN / 128, D_MODEL / 64);
    gemm_bf16x3<<<ggrid, 256, 0, stream>>>(x, Wq, bq,      qws,  0);
    gemm_bf16x3<<<ggrid, 256, 0, stream>>>(x, Wk, nullptr, kseg, 1);
    gemm_bf16x3<<<ggrid, 256, 0, stream>>>(x, Wv, bv,      vseg, 1);

    dim3 agrid(TLEN / 128, N_HEAD, BATCH);
    attn_flash<<<agrid, 256, 0, stream>>>(qws, kseg, vseg, attnws);

    gemm_bf16x3<<<ggrid, 256, 0, stream>>>(attnws, Wo, bo, outp, 0);
}
